// CrossAttentionMoEPolicy_82222853915170
// MI455X (gfx1250) — matrix operand order repaired and checked by reading, not compiled, not run on hardware
//
#include <hip/hip_runtime.h>
#include <hip/hip_bf16.h>

// ---------------------------------------------------------------------------
// Fully-fused CrossAttention+MoE policy net for gfx1250 (MI455X).
// All GEMMs run on v_wmma_f32_16x16x32_f16 (fp16 in, fp32 accumulate).
// One wave owns 16 rows end-to-end; no inter-wave synchronization needed.
// ---------------------------------------------------------------------------

typedef __attribute__((ext_vector_type(16))) _Float16 v16h;
typedef __attribute__((ext_vector_type(8)))  _Float16 v8h;
typedef __attribute__((ext_vector_type(4)))  _Float16 v4h;
typedef __attribute__((ext_vector_type(8)))  float    v8f;

#define BTOT 131072

// f16 weight image offsets in d_ws (units: halves; all multiples of 1024 -> 32B aligned)
constexpr size_t OFF_VE_W1 = 0;          // 256 x 128   (K padded 115->128)
constexpr size_t OFF_VE_W2 = 32768;      // 256 x 256
constexpr size_t OFF_WQ    = 98304;      // 256 x 256
constexpr size_t OFF_WK    = 163840;     // 256 x 256
constexpr size_t OFF_WV    = 229376;     // 256 x 256
constexpr size_t OFF_WO    = 294912;     // 256 x 256
constexpr size_t OFF_RW1   = 360448;     // 128 x 256
constexpr size_t OFF_RW2   = 393216;     //  64 x 128
constexpr size_t OFF_RW3   = 401408;     //  16 x 64   (rows padded 4->16)
constexpr size_t OFF_EW1   = 402432;     // 4 x 128 x 256
constexpr size_t OFF_EW2   = 533504;     // 4 x 128 x 128
constexpr size_t OFF_EW3   = 599040;     // 4 x  32 x 128 (rows padded 23->32)
constexpr size_t OFF_CW1   = 615424;     // 128 x 256
constexpr size_t OFF_CW2   = 648192;     //  64 x 128
constexpr size_t OFF_CW3   = 656384;     //  16 x 64   (rows padded 1->16)

// output layout: policy(B*23) | value(B) | alpha(B*4) | c(B*256)
constexpr size_t OUT_VAL   = (size_t)BTOT * 23;
constexpr size_t OUT_ALPHA = (size_t)BTOT * 24;
constexpr size_t OUT_C     = (size_t)BTOT * 28;

constexpr int SMEM_BYTES = 4 * 64 * 256 * 2;   // four f16 activation buffers = 128 KB

struct Params {
  const float* state; const float* hlogic;
  const _Float16* wf;
  const float *ve_b1,*ve_g1,*ve_be1,*ve_b2,*ve_g2,*ve_be2;
  const float *bq,*bk,*bv,*bo,*ca_g,*ca_b;
  const float *r_b1,*r_g1,*r_be1,*r_b2,*r_g2,*r_be2,*r_b3;
  const float *e_b1,*e_g1,*e_be1,*e_b2,*e_g2,*e_be2,*e_b3;
  const float *c_b1,*c_g1,*c_be1,*c_b2,*c_g2,*c_be2,*c_b3;
  float* out;
};

// ---- weight prep: f32 -> f16 with zero padding --------------------------------
__global__ void cvt_pad_f16(const float* __restrict__ src, _Float16* __restrict__ dst,
                            int srcRows, int srcCols, int dstRows, int dstCols) {
  int i = blockIdx.x * blockDim.x + threadIdx.x;
  int total = dstRows * dstCols;
  if (i >= total) return;
  int r = i / dstCols, c = i - r * dstCols;
  float v = (r < srcRows && c < srcCols) ? src[(size_t)r * srcCols + c] : 0.0f;
  dst[i] = (_Float16)v;
}

// ---- WMMA tile GEMM: this wave's 16 rows (A from LDS) x W^T (f16, global) ------
// A layout (16x32 f16): lane m in [0,16) rows, 16 contiguous K halves (lanes 16-31: K 16-31)
// B layout (32x16 f16): lane n cols, 16 contiguous K halves (lanes 16-31: K 16-31)
template<int NT>
__device__ inline void gemm(const _Float16* A0, int lda,
                            const _Float16* __restrict__ W, int kp,
                            v8f* acc, int lane) {
  const int m    = lane & 15;
  const int aoff = (lane >> 4) * 16;
  const int boff = (lane >> 4) * 16;
  const v8f zero = {0.f,0.f,0.f,0.f,0.f,0.f,0.f,0.f};
#pragma unroll
  for (int nt = 0; nt < NT; ++nt) acc[nt] = zero;
  const int nk = kp >> 5;
  for (int kt = 0; kt < nk; ++kt) {
    const _Float16* pa = A0 + m * lda + (kt << 5) + aoff;
    v8h alo = *(const v8h*)pa;
    v8h ahi = *(const v8h*)(pa + 8);
    v16h a = __builtin_shufflevector(alo, ahi, 0,1,2,3,4,5,6,7,8,9,10,11,12,13,14,15);
#pragma unroll
    for (int nt = 0; nt < NT; ++nt) {
      const _Float16* pb = W + (size_t)(nt * 16 + m) * kp + (kt << 5) + boff;
      v16h b = *(const v16h*)pb;
      acc[nt] = __builtin_amdgcn_wmma_f32_16x16x32_f16(false, a, false, b,
                                                       (short)0, acc[nt], false, false);
    }
  }
}

// ---- bias (+residual) (+ReLU) + LayerNorm epilogue; stats via half-wave shuffles
template<int NT>
__device__ inline void epi_ln(v8f* acc, const float* bias, const float* g, const float* be,
                              const _Float16* resid, _Float16* dst, float* gout,
                              int lane, bool relu) {
  const int n15 = lane & 15;
  const int rhalf = (lane >> 4) << 3;
  float sum[8], sq[8];
#pragma unroll
  for (int r = 0; r < 8; ++r) { sum[r] = 0.f; sq[r] = 0.f; }
#pragma unroll
  for (int nt = 0; nt < NT; ++nt) {
    const int col = nt * 16 + n15;
    const float b = bias[col];
#pragma unroll
    for (int r = 0; r < 8; ++r) {
      float v = acc[nt][r] + b;
      if (resid) v += (float)resid[(r + rhalf) * 256 + col];
      if (relu)  v = fmaxf(v, 0.f);
      acc[nt][r] = v;
      sum[r] += v; sq[r] += v * v;
    }
  }
#pragma unroll
  for (int mk = 1; mk < 16; mk <<= 1) {
#pragma unroll
    for (int r = 0; r < 8; ++r) {
      sum[r] += __shfl_xor(sum[r], mk, 32);
      sq[r]  += __shfl_xor(sq[r],  mk, 32);
    }
  }
  const float inv = 1.0f / (float)(NT * 16);
#pragma unroll
  for (int r = 0; r < 8; ++r) {
    float mean = sum[r] * inv;
    float var  = sq[r] * inv - mean * mean;
    sum[r] = mean;
    sq[r]  = rsqrtf(var + 1e-5f);
  }
#pragma unroll
  for (int nt = 0; nt < NT; ++nt) {
    const int col = nt * 16 + n15;
    const float gg = g[col], bb = be[col];
#pragma unroll
    for (int r = 0; r < 8; ++r) {
      float v = (acc[nt][r] - sum[r]) * sq[r] * gg + bb;
      dst[(r + rhalf) * 256 + col] = (_Float16)v;
      if (gout) gout[(size_t)(r + rhalf) * 256 + col] = v;
    }
  }
}

__device__ inline float nan0(float x) {
  if (x != x) return 0.f;
  if (x >  3.0e38f) return  3.4028235e38f;
  if (x < -3.0e38f) return -3.4028235e38f;
  return x;
}

// ---- fused forward kernel -----------------------------------------------------
__global__ __launch_bounds__(128) void policy_fused(Params P) {
  extern __shared__ _Float16 smem[];
  const int lane = threadIdx.x & 31;
  const int wave = threadIdx.x >> 5;
  const int rowbase = blockIdx.x * 64 + wave * 16;
  const int n15 = lane & 15;
  const int rhalf = (lane >> 4) << 3;

  _Float16* bufL = smem;                 // h_logic (residual + Q input)
  _Float16* bufC = bufL + 64 * 256;      // c (shared input of router/experts/critic)
  _Float16* bufX = bufC + 64 * 256;      // ping
  _Float16* bufY = bufX + 64 * 256;      // pong

  _Float16* L = bufL + wave * 16 * 256;
  _Float16* C = bufC + wave * 16 * 256;
  _Float16* X = bufX + wave * 16 * 256;
  _Float16* Y = bufY + wave * 16 * 256;

  // stage-in: state (padded 115->128, rows only 4B-aligned -> scalar loads)
  for (int i = lane; i < 16 * 128; i += 32) {
    int r = i >> 7, c = i & 127;
    float v = (c < 115) ? P.state[(size_t)(rowbase + r) * 115 + c] : 0.f;
    X[r * 256 + c] = (_Float16)v;
  }
  // h_logic rows are 256 f32 (16B aligned): float4 loads + 8B LDS stores
  for (int i = lane; i < 16 * 64; i += 32) {
    int r = i >> 6, c4 = i & 63;
    float4 v = *(const float4*)(P.hlogic + (size_t)(rowbase + r) * 256 + c4 * 4);
    v4h h = { (_Float16)v.x, (_Float16)v.y, (_Float16)v.z, (_Float16)v.w };
    *(v4h*)&L[r * 256 + c4 * 4] = h;
  }

  v8f acc[16];

  // visual encoder
  gemm<16>(X, 256, P.wf + OFF_VE_W1, 128, acc, lane);
  epi_ln<16>(acc, P.ve_b1, P.ve_g1, P.ve_be1, nullptr, Y, nullptr, lane, true);
  gemm<16>(Y, 256, P.wf + OFF_VE_W2, 256, acc, lane);
  epi_ln<16>(acc, P.ve_b2, P.ve_g2, P.ve_be2, nullptr, X, nullptr, lane, true);  // X = h_vis

  // Q = h_logic @ wq^T + bq  -> Y
  gemm<16>(L, 256, P.wf + OFF_WQ, 256, acc, lane);
#pragma unroll
  for (int nt = 0; nt < 16; ++nt) {
    float b = P.bq[nt * 16 + n15];
#pragma unroll
    for (int r = 0; r < 8; ++r)
      Y[(r + rhalf) * 256 + nt * 16 + n15] = (_Float16)(acc[nt][r] + b);
  }

  // K stage + per-head scores (dot with Q in LDS); top-1 head -> one-hot attention
  gemm<16>(X, 256, P.wf + OFF_WK, 256, acc, lane);
  float dot[32];
#pragma unroll
  for (int i = 0; i < 32; ++i) dot[i] = 0.f;
#pragma unroll
  for (int nt = 0; nt < 16; ++nt) {
    float b = P.bk[nt * 16 + n15];
    int h = nt >> 2;
#pragma unroll
    for (int r = 0; r < 8; ++r) {
      float kv = acc[nt][r] + b;
      float qv = (float)Y[(r + rhalf) * 256 + nt * 16 + n15];
      dot[r * 4 + h] += kv * qv;
    }
  }
#pragma unroll
  for (int mk = 1; mk < 16; mk <<= 1)
#pragma unroll
    for (int i = 0; i < 32; ++i) dot[i] += __shfl_xor(dot[i], mk, 32);
  int hstar[8];
#pragma unroll
  for (int r = 0; r < 8; ++r) {
    float best = -3.0e38f; int bi = 0;
#pragma unroll
    for (int h = 0; h < 4; ++h) {
      float s = dot[r * 4 + h] * 0.125f;            // / sqrt(64)
      if (s > best) { best = s; bi = h; }           // ties -> lowest index (top_k)
    }
    hstar[r] = bi;
  }

  // V stage, masked to selected head -> Y (context)
  gemm<16>(X, 256, P.wf + OFF_WV, 256, acc, lane);
#pragma unroll
  for (int nt = 0; nt < 16; ++nt) {
    float b = P.bv[nt * 16 + n15];
    int h = nt >> 2;
#pragma unroll
    for (int r = 0; r < 8; ++r) {
      float v = acc[nt][r] + b;
      v = (h == hstar[r]) ? v : 0.f;
      Y[(r + rhalf) * 256 + nt * 16 + n15] = (_Float16)v;
    }
  }

  // c = LN(context @ wo^T + bo + h_logic); also stream c to global output
  gemm<16>(Y, 256, P.wf + OFF_WO, 256, acc, lane);
  epi_ln<16>(acc, P.bo, P.ca_g, P.ca_b, L, C,
             P.out + OUT_C + (size_t)rowbase * 256, lane, false);

  // router
  gemm<8>(C, 256, P.wf + OFF_RW1, 256, acc, lane);
  epi_ln<8>(acc, P.r_b1, P.r_g1, P.r_be1, nullptr, X, nullptr, lane, true);
  gemm<4>(X, 256, P.wf + OFF_RW2, 128, acc, lane);
  epi_ln<4>(acc, P.r_b2, P.r_g2, P.r_be2, nullptr, Y, nullptr, lane, true);
  gemm<1>(Y, 256, P.wf + OFF_RW3, 64, acc, lane);

  // router logits -> top-2 softmax scatter; alpha kept in registers (per half-wave)
  float alphaReg[8];
  const int src = lane & 16;   // base lane of this half-wave
  {
    float badd = (n15 < 4) ? P.r_b3[n15] : 0.f;
#pragma unroll
    for (int r = 0; r < 8; ++r) acc[0][r] += badd;
#pragma unroll
    for (int r = 0; r < 8; ++r) {
      float l[4];
      l[0] = nan0(__shfl(acc[0][r], src + 0, 32));
      l[1] = nan0(__shfl(acc[0][r], src + 1, 32));
      l[2] = nan0(__shfl(acc[0][r], src + 2, 32));
      l[3] = nan0(__shfl(acc[0][r], src + 3, 32));
      int i1 = 0; float v1 = l[0];
#pragma unroll
      for (int j = 1; j < 4; ++j) if (l[j] > v1) { v1 = l[j]; i1 = j; }
      int i2 = -1; float v2 = -3.0e38f;
#pragma unroll
      for (int j = 0; j < 4; ++j) if (j != i1 && l[j] > v2) { v2 = l[j]; i2 = j; }
      float e2 = __expf(v2 - v1);
      float p1 = 1.f / (1.f + e2);
      float p2 = e2 / (1.f + e2);
      float a = (n15 == i1) ? p1 : (n15 == i2) ? p2 : 0.f;
      alphaReg[r] = a;          // lane src+j holds alpha[row r+rhalf][j] for j<4
      if (n15 < 4)
        P.out[OUT_ALPHA + (size_t)(rowbase + r + rhalf) * 4 + n15] = a;
    }
  }

  // experts (dense); alpha-weighted mix accumulated in registers
  float pol[2][8];
#pragma unroll
  for (int nt = 0; nt < 2; ++nt)
#pragma unroll
    for (int r = 0; r < 8; ++r) pol[nt][r] = 0.f;

  for (int e = 0; e < 4; ++e) {
    gemm<8>(C, 256, P.wf + OFF_EW1 + (size_t)e * 128 * 256, 256, acc, lane);
    epi_ln<8>(acc, P.e_b1 + e * 128, P.e_g1 + e * 128, P.e_be1 + e * 128,
              nullptr, X, nullptr, lane, true);
    gemm<8>(X, 256, P.wf + OFF_EW2 + (size_t)e * 128 * 128, 128, acc, lane);
    epi_ln<8>(acc, P.e_b2 + e * 128, P.e_g2 + e * 128, P.e_be2 + e * 128,
              nullptr, Y, nullptr, lane, true);
    gemm<2>(Y, 256, P.wf + OFF_EW3 + (size_t)e * 32 * 128, 128, acc, lane);
    const float* b3 = P.e_b3 + e * 23;
#pragma unroll
    for (int nt = 0; nt < 2; ++nt) {
      int col = nt * 16 + n15;
      float bb = (col < 23) ? b3[col] : 0.f;
#pragma unroll
      for (int r = 0; r < 8; ++r) {
        float ae = __shfl(alphaReg[r], src + e, 32);   // alpha[row][e] broadcast
        pol[nt][r] += ae * (acc[nt][r] + bb);
      }
    }
  }

  // critic
  gemm<8>(C, 256, P.wf + OFF_CW1, 256, acc, lane);
  epi_ln<8>(acc, P.c_b1, P.c_g1, P.c_be1, nullptr, X, nullptr, lane, true);
  gemm<4>(X, 256, P.wf + OFF_CW2, 128, acc, lane);
  epi_ln<4>(acc, P.c_b2, P.c_g2, P.c_be2, nullptr, Y, nullptr, lane, true);
  gemm<1>(Y, 256, P.wf + OFF_CW3, 64, acc, lane);
  if (n15 == 0) {
#pragma unroll
    for (int r = 0; r < 8; ++r)
      P.out[OUT_VAL + (size_t)(rowbase + r + rhalf)] = acc[0][r] + P.c_b3[0];
  }

  // policy write-out straight from registers
#pragma unroll
  for (int nt = 0; nt < 2; ++nt) {
    int col = nt * 16 + n15;
    if (col < 23) {
#pragma unroll
      for (int r = 0; r < 8; ++r)
        P.out[(size_t)(rowbase + r + rhalf) * 23 + col] = pol[nt][r];
    }
  }
}

// ---------------------------------------------------------------------------
extern "C" void kernel_launch(void* const* d_in, const int* in_sizes, int n_in,
                              void* d_out, int out_size, void* d_ws, size_t ws_size,
                              hipStream_t stream) {
  (void)in_sizes; (void)n_in; (void)out_size; (void)ws_size;
  _Float16* wf = (_Float16*)d_ws;
  auto F = [&](int i) { return (const float*)d_in[i]; };
  auto cvt = [&](int i, size_t off, int sr, int sc, int dr, int dc, size_t srcElemOff = 0) {
    int total = dr * dc;
    cvt_pad_f16<<<(total + 255) / 256, 256, 0, stream>>>(F(i) + srcElemOff, wf + off,
                                                         sr, sc, dr, dc);
  };
  // weight prep (f32 -> padded f16 in workspace)
  cvt(2,  OFF_VE_W1, 256, 115, 256, 128);
  cvt(6,  OFF_VE_W2, 256, 256, 256, 256);
  cvt(10, OFF_WQ,    256, 256, 256, 256);
  cvt(12, OFF_WK,    256, 256, 256, 256);
  cvt(14, OFF_WV,    256, 256, 256, 256);
  cvt(16, OFF_WO,    256, 256, 256, 256);
  cvt(20, OFF_RW1,   128, 256, 128, 256);
  cvt(24, OFF_RW2,    64, 128,  64, 128);
  cvt(28, OFF_RW3,     4,  64,  16,  64);
  cvt(30, OFF_EW1,   512, 256, 512, 256);   // 4 experts x 128 rows
  cvt(34, OFF_EW2,   512, 128, 512, 128);
  for (int e = 0; e < 4; ++e)
    cvt(38, OFF_EW3 + (size_t)e * 32 * 128, 23, 128, 32, 128, (size_t)e * 23 * 128);
  cvt(40, OFF_CW1,   128, 256, 128, 256);
  cvt(44, OFF_CW2,    64, 128,  64, 128);
  cvt(48, OFF_CW3,     1,  64,  16,  64);

  Params P;
  P.state = F(0); P.hlogic = F(1); P.wf = wf;
  P.ve_b1 = F(3);  P.ve_g1 = F(4);  P.ve_be1 = F(5);
  P.ve_b2 = F(7);  P.ve_g2 = F(8);  P.ve_be2 = F(9);
  P.bq = F(11); P.bk = F(13); P.bv = F(15); P.bo = F(17); P.ca_g = F(18); P.ca_b = F(19);
  P.r_b1 = F(21); P.r_g1 = F(22); P.r_be1 = F(23);
  P.r_b2 = F(25); P.r_g2 = F(26); P.r_be2 = F(27); P.r_b3 = F(29);
  P.e_b1 = F(31); P.e_g1 = F(32); P.e_be1 = F(33);
  P.e_b2 = F(35); P.e_g2 = F(36); P.e_be2 = F(37); P.e_b3 = F(39);
  P.c_b1 = F(41); P.c_g1 = F(42); P.c_be1 = F(43);
  P.c_b2 = F(45); P.c_g2 = F(46); P.c_be2 = F(47); P.c_b3 = F(49);
  P.out = (float*)d_out;

  policy_fused<<<BTOT / 64, 128, SMEM_BYTES, stream>>>(P);
}
